// EnergyConstrainedPredictiveCodingModel_77910706749637
// MI455X (gfx1250) — compile-verified, matrix operand order checked by reading, not run on hardware
//
#include <hip/hip_runtime.h>
#include <hip/hip_bf16.h>

typedef __attribute__((ext_vector_type(16))) _Float16 v16h;
typedef __attribute__((ext_vector_type(8)))  float    v8f;
typedef __attribute__((ext_vector_type(4))) unsigned int u32x4;
typedef __attribute__((ext_vector_type(8))) int          i32x8;
typedef __attribute__((ext_vector_type(4))) int          i32x4;

#define WMMA_F16(a, b, c) \
  __builtin_amdgcn_wmma_f32_16x16x32_f16(false, (a), false, (b), (short)0, (c), false, false)

// Use the Tensor Data Mover when the toolchain exposes it; otherwise fall back
// to the plain global-load path (compile can't regress).
#if defined(__HIP_DEVICE_COMPILE__) && defined(__has_builtin)
# if __has_builtin(__builtin_amdgcn_tensor_load_to_lds) && \
     __has_builtin(__builtin_amdgcn_s_wait_tensorcnt)
#  define USE_TDM 1
# endif
#endif
#ifndef USE_TDM
# define USE_TDM 0
#endif

// ---------------------------------------------------------------------------
// Problem constants
// ---------------------------------------------------------------------------
#define BATCH   8192
#define D_IN    1024
#define D_Z     256
#define D_H     256
#define D_T     16
#define D_REC   256

// ---------------------------------------------------------------------------
// Helpers
// ---------------------------------------------------------------------------
__device__ __forceinline__ v16h ldT(const _Float16* __restrict__ base, int ld,
                                    int row0, int k0, int lane) {
  // lanes 0..15: rows 0..15, K-chunk [k0, k0+16); lanes 16..31: K-chunk [k0+16, k0+32)
  const _Float16* p = base + (size_t)(row0 + (lane & 15)) * (size_t)ld
                           + (size_t)(k0 + ((lane >> 4) << 4));
  return *(const v16h*)p;
}

__device__ __forceinline__ float softplus_b(float x, float beta, float inv_beta) {
  float bx = beta * x;
  if (bx > 20.f) return x;
  return logf(1.f + __expf(bx)) * inv_beta;
}

__device__ __forceinline__ float wave_sum(float v) {
#pragma unroll
  for (int o = 16; o > 0; o >>= 1) v += __shfl_xor(v, o, 32);
  return v;
}

#if USE_TDM
// Issue a TDM DMA of a 2D row-major panel (tileRows x tileCols f16 elements,
// row stride = strideElems) from global memory into LDS at byte offset ldsOff.
// D# layout per CDNA5 ISA.
__device__ __forceinline__ void tdm_load_panel(unsigned ldsOff,
                                               const _Float16* gsrc,
                                               unsigned tileCols, unsigned tileRows,
                                               unsigned strideElems) {
  unsigned long long ga = (unsigned long long)(uintptr_t)gsrc;
  u32x4 g0;
  g0[0] = 1u;                                   // count=1 (valid), user descriptor
  g0[1] = ldsOff;                               // LDS byte address
  g0[2] = (unsigned)(ga & 0xFFFFFFFFu);         // global_addr[31:0]
  g0[3] = (unsigned)((ga >> 32) & 0x01FFFFFFu)  // global_addr[56:32]
          | (2u << 30);                         // type=2 ("image")
  i32x8 g1;
  g1[0] = (int)(1u << 16);                      // wg_mask=0, data_size=1 (2 bytes)
  g1[1] = (int)((tileCols & 0xFFFFu) << 16);    // tensor_dim0[15:0] @bits[63:48]
  g1[2] = (int)((tileCols >> 16) | ((tileRows & 0xFFFFu) << 16)); // td0 hi | td1 lo
  g1[3] = (int)((tileRows >> 16) | ((tileCols & 0xFFFFu) << 16)); // td1 hi | tile_dim0
  g1[4] = (int)(tileRows & 0xFFFFu);            // tile_dim1 | tile_dim2=0
  g1[5] = (int)strideElems;                     // tensor_dim0_stride[31:0]
  g1[6] = 0;                                    // stride0 hi | stride1 lo
  g1[7] = 0;                                    // stride1 hi
  i32x4 z4 = {0, 0, 0, 0};
#if __clang_major__ >= 23
  i32x8 z8 = {0, 0, 0, 0, 0, 0, 0, 0};
  __builtin_amdgcn_tensor_load_to_lds(g0, g1, z4, z4, z8, 0);
#else
  __builtin_amdgcn_tensor_load_to_lds(g0, g1, z4, z4, 0);
#endif
}
#endif

// ---------------------------------------------------------------------------
// Conversion kernels
// ---------------------------------------------------------------------------
__global__ void cvt_f16_kernel(const float* __restrict__ s, _Float16* __restrict__ d,
                               int n, int doRelu) {
  int i = blockIdx.x * blockDim.x + threadIdx.x;
  if (i < n) {
    float v = s[i];
    if (doRelu) v = fmaxf(v, 0.f);
    d[i] = (_Float16)v;
  }
}

// relu(W_t2z) [D_Z,16] -> zero-padded f16 [D_Z,32]
__global__ void t2z_pad_kernel(const float* __restrict__ s, _Float16* __restrict__ d) {
  int i = blockIdx.x * blockDim.x + threadIdx.x;
  if (i < D_Z * 32) {
    int r = i >> 5, c = i & 31;
    float v = (c < D_T) ? fmaxf(s[r * D_T + c], 0.f) : 0.f;
    d[i] = (_Float16)v;
  }
}

// ---------------------------------------------------------------------------
// K1: prior pathway (1M x 2N tiles per wave).
// ---------------------------------------------------------------------------
__global__ void prior_kernel(const _Float16* __restrict__ Ah,
                             const _Float16* __restrict__ Wmu,
                             const _Float16* __restrict__ Wlv,
                             const float* __restrict__ eps_zhat,
                             float* __restrict__ zhat,
                             _Float16* __restrict__ sigp_h) {
  const int lane = threadIdx.x & 31;
  int g = blockIdx.x * (blockDim.x >> 5) + (threadIdx.x >> 5);
  const int mTiles = BATCH / 16;
  int m0 = (g % mTiles) << 4;
  int n0 = (g / mTiles) << 5;
  v8f aMu0 = {}, aMu1 = {}, aLv0 = {}, aLv1 = {};
  for (int k = 0; k < D_H; k += 32) {
    v16h a    = ldT(Ah,  D_H, m0,      k, lane);
    v16h bmu0 = ldT(Wmu, D_H, n0,      k, lane);
    v16h bmu1 = ldT(Wmu, D_H, n0 + 16, k, lane);
    v16h blv0 = ldT(Wlv, D_H, n0,      k, lane);
    v16h blv1 = ldT(Wlv, D_H, n0 + 16, k, lane);
    aMu0 = WMMA_F16(a, bmu0, aMu0);
    aMu1 = WMMA_F16(a, bmu1, aMu1);
    aLv0 = WMMA_F16(a, blv0, aLv0);
    aLv1 = WMMA_F16(a, blv1, aLv1);
  }
#pragma unroll
  for (int c = 0; c < 2; ++c) {
    v8f aM = c ? aMu1 : aMu0;
    v8f aL = c ? aLv1 : aLv0;
    int n = n0 + (c << 4) + (lane & 15);
#pragma unroll
    for (int j = 0; j < 8; ++j) {
      int m = m0 + j + ((lane >> 4) << 3);
      int idx = m * D_Z + n;
      float mu = fmaxf(aM[j], 0.f);
      float sg = softplus_b(aL[j], 1.2f, 1.f / 1.2f);
      zhat[idx] = mu + eps_zhat[idx] * sg;
      sigp_h[idx] = (_Float16)sg;
    }
  }
}

// ---------------------------------------------------------------------------
// K2: h = relu(z_m_1@Wz2h^T + h_m_1@Wh2h^T)  (1M x 2N)
// ---------------------------------------------------------------------------
__global__ void hstate_kernel(const _Float16* __restrict__ Zm1,
                              const _Float16* __restrict__ Hm1,
                              const _Float16* __restrict__ Wz2h,
                              const _Float16* __restrict__ Wh2h,
                              float* __restrict__ hOut) {
  const int lane = threadIdx.x & 31;
  int g = blockIdx.x * (blockDim.x >> 5) + (threadIdx.x >> 5);
  const int mTiles = BATCH / 16;
  int m0 = (g % mTiles) << 4;
  int n0 = (g / mTiles) << 5;
  v8f acc0 = {}, acc1 = {};
  for (int k = 0; k < D_Z; k += 32) {
    v16h az = ldT(Zm1,  D_Z, m0,      k, lane);
    v16h b0 = ldT(Wz2h, D_Z, n0,      k, lane);
    v16h b1 = ldT(Wz2h, D_Z, n0 + 16, k, lane);
    acc0 = WMMA_F16(az, b0, acc0);
    acc1 = WMMA_F16(az, b1, acc1);
  }
  for (int k = 0; k < D_H; k += 32) {
    v16h ah = ldT(Hm1,  D_H, m0,      k, lane);
    v16h b0 = ldT(Wh2h, D_H, n0,      k, lane);
    v16h b1 = ldT(Wh2h, D_H, n0 + 16, k, lane);
    acc0 = WMMA_F16(ah, b0, acc0);
    acc1 = WMMA_F16(ah, b1, acc1);
  }
#pragma unroll
  for (int c = 0; c < 2; ++c) {
    v8f acc = c ? acc1 : acc0;
    int n = n0 + (c << 4) + (lane & 15);
#pragma unroll
    for (int j = 0; j < 8; ++j) {
      int m = m0 + j + ((lane >> 4) << 3);
      hOut[m * D_H + n] = fmaxf(acc[j], 0.f);
    }
  }
}

// ---------------------------------------------------------------------------
// K3: theta pathway
// ---------------------------------------------------------------------------
__global__ void theta_kernel(const _Float16* __restrict__ Ith,
                             const _Float16* __restrict__ Wi2t,
                             const _Float16* __restrict__ Sigph,
                             const _Float16* __restrict__ Wvip,
                             const float* __restrict__ theta_m1,
                             _Float16* __restrict__ theta16) {
  const int lane = threadIdx.x & 31;
  int g = blockIdx.x * (blockDim.x >> 5) + (threadIdx.x >> 5);
  int m0 = g << 4;
  v8f acc1 = {}, acc2 = {};
  for (int k = 0; k < D_IN; k += 32) {
    v16h a = ldT(Ith,  D_IN, m0, k, lane);
    v16h b = ldT(Wi2t, D_IN, 0,  k, lane);
    acc1 = WMMA_F16(a, b, acc1);
  }
  for (int k = 0; k < D_Z; k += 32) {
    v16h a = ldT(Sigph, D_Z, m0, k, lane);
    v16h b = ldT(Wvip,  D_Z, 0,  k, lane);
    acc2 = WMMA_F16(a, b, acc2);
  }
  int n = lane & 15;
#pragma unroll
  for (int j = 0; j < 8; ++j) {
    int m = m0 + j + ((lane >> 4) << 3);
    float th = 0.5f * theta_m1[m * D_T + n] + 0.1f * acc1[j] - acc2[j];
    float tv = 0.001f * softplus_b(th, 0.5f, 2.f);
    theta16[m * 32 + n]      = (_Float16)tv;
    theta16[m * 32 + 16 + n] = (_Float16)0.f;   // zero K-padding
  }
}

// ---------------------------------------------------------------------------
// K4: posterior + z (1M x 2N), temporal & energy loss partial sums.
//     B panels (Wpm/Wpl rows n0..n0+31, full K=1024) staged into LDS by the
//     Tensor Data Mover once per workgroup, consumed by all 8 waves.
// ---------------------------------------------------------------------------
__global__ void posterior_kernel(const _Float16* __restrict__ Ith,
                                 const _Float16* __restrict__ Wpm,
                                 const _Float16* __restrict__ Wpl,
                                 const _Float16* __restrict__ theta16,
                                 const _Float16* __restrict__ Wt2z,   // [D_Z,32] padded
                                 const float* __restrict__ eps_z,
                                 const float* __restrict__ zhat,
                                 float* __restrict__ zOut,
                                 _Float16* __restrict__ zH,
                                 float* __restrict__ loss) {
  extern __shared__ _Float16 ldsW[];   // 2 * 32 * 1024 f16 = 128 KB
  const int lane = threadIdx.x & 31;
  const int wave = threadIdx.x >> 5;
  const int mTiles = BATCH / 16;
  int gBlock = blockIdx.x * (blockDim.x >> 5);
  int n0 = (gBlock / mTiles) << 5;               // block-uniform N group
  int m0 = ((gBlock + wave) % mTiles) << 4;

#if USE_TDM
  if (wave == 0) {
    unsigned ldsBase = (unsigned)(uintptr_t)(void*)ldsW;   // low 32 bits = LDS byte offset
    tdm_load_panel(ldsBase,                    Wpm + (size_t)n0 * D_IN, D_IN, 32, D_IN);
    tdm_load_panel(ldsBase + 32u * D_IN * 2u,  Wpl + (size_t)n0 * D_IN, D_IN, 32, D_IN);
    __builtin_amdgcn_s_wait_tensorcnt((short)0);
  }
  __syncthreads();
  const _Float16* panM = ldsW;                 // rows n0..n0+31 of Wpm
  const _Float16* panL = ldsW + 32 * D_IN;     // rows n0..n0+31 of Wpl
  const int bn0 = 0, bn1 = 16;
#else
  const _Float16* panM = Wpm;
  const _Float16* panL = Wpl;
  const int bn0 = n0, bn1 = n0 + 16;
#endif

  v8f aM0 = {}, aM1 = {}, aL0 = {}, aL1 = {}, aT0 = {}, aT1 = {};
  for (int k = 0; k < D_IN; k += 32) {
    v16h a   = ldT(Ith,  D_IN, m0,  k, lane);
    v16h bm0 = ldT(panM, D_IN, bn0, k, lane);
    v16h bm1 = ldT(panM, D_IN, bn1, k, lane);
    v16h bl0 = ldT(panL, D_IN, bn0, k, lane);
    v16h bl1 = ldT(panL, D_IN, bn1, k, lane);
    aM0 = WMMA_F16(a, bm0, aM0);
    aM1 = WMMA_F16(a, bm1, aM1);
    aL0 = WMMA_F16(a, bl0, aL0);
    aL1 = WMMA_F16(a, bl1, aL1);
  }
  {
    v16h a  = ldT(theta16, 32, m0,      0, lane);
    v16h b0 = ldT(Wt2z,    32, n0,      0, lane);
    v16h b1 = ldT(Wt2z,    32, n0 + 16, 0, lane);
    aT0 = WMMA_F16(a, b0, aT0);
    aT1 = WMMA_F16(a, b1, aT1);
  }
  float esum = 0.f, tsum = 0.f;
#pragma unroll
  for (int c = 0; c < 2; ++c) {
    v8f accM = c ? aM1 : aM0;
    v8f accL = c ? aL1 : aL0;
    v8f accT = c ? aT1 : aT0;
    int n = n0 + (c << 4) + (lane & 15);
#pragma unroll
    for (int j = 0; j < 8; ++j) {
      int m = m0 + j + ((lane >> 4) << 3);
      int idx = m * D_Z + n;
      float mu = fmaxf(accM[j], 0.f);
      float sq = fmaxf(accL[j], 0.f);
      float raw = fminf(fmaxf(mu + eps_z[idx] * sq, 0.f), 1.f);
      float zv = fmaxf(raw - 10.f * accT[j], 0.f);
      zOut[idx] = zv;
      zH[idx] = (_Float16)zv;
      esum += zv;                         // |z| == z (z >= 0)
      float d = zv - zhat[idx];
      tsum += d * d;
    }
  }
  esum = wave_sum(esum);
  tsum = wave_sum(tsum);
  if (lane == 0) {
    atomicAdd(&loss[1], tsum);
    atomicAdd(&loss[2], esum);
  }
}

// ---------------------------------------------------------------------------
// K5: zr = z @ W_rec1^T  (1M x 2N, f16 out)
// ---------------------------------------------------------------------------
__global__ void rec1_kernel(const _Float16* __restrict__ Zh,
                            const _Float16* __restrict__ Wr1,
                            _Float16* __restrict__ ZrH) {
  const int lane = threadIdx.x & 31;
  int g = blockIdx.x * (blockDim.x >> 5) + (threadIdx.x >> 5);
  const int mTiles = BATCH / 16;
  int m0 = (g % mTiles) << 4;
  int n0 = (g / mTiles) << 5;
  v8f acc0 = {}, acc1 = {};
  for (int k = 0; k < D_Z; k += 32) {
    v16h a  = ldT(Zh,  D_Z, m0,      k, lane);
    v16h b0 = ldT(Wr1, D_Z, n0,      k, lane);
    v16h b1 = ldT(Wr1, D_Z, n0 + 16, k, lane);
    acc0 = WMMA_F16(a, b0, acc0);
    acc1 = WMMA_F16(a, b1, acc1);
  }
#pragma unroll
  for (int c = 0; c < 2; ++c) {
    v8f acc = c ? acc1 : acc0;
    int n = n0 + (c << 4) + (lane & 15);
#pragma unroll
    for (int j = 0; j < 8; ++j) {
      int m = m0 + j + ((lane >> 4) << 3);
      ZrH[m * D_REC + n] = (_Float16)acc[j];
    }
  }
}

// ---------------------------------------------------------------------------
// K6: I_hat = sigmoid(zr @ W_rec2^T) (1M x 2N), spatial loss partial sums
// ---------------------------------------------------------------------------
__global__ void rec2_kernel(const _Float16* __restrict__ ZrH,
                            const _Float16* __restrict__ Wr2,
                            const float* __restrict__ It,
                            float* __restrict__ IhatOut,
                            float* __restrict__ loss) {
  const int lane = threadIdx.x & 31;
  int g = blockIdx.x * (blockDim.x >> 5) + (threadIdx.x >> 5);
  const int mTiles = BATCH / 16;
  int m0 = (g % mTiles) << 4;
  int n0 = (g / mTiles) << 5;
  v8f acc0 = {}, acc1 = {};
  for (int k = 0; k < D_REC; k += 32) {
    v16h a  = ldT(ZrH, D_REC, m0,      k, lane);
    v16h b0 = ldT(Wr2, D_REC, n0,      k, lane);
    v16h b1 = ldT(Wr2, D_REC, n0 + 16, k, lane);
    acc0 = WMMA_F16(a, b0, acc0);
    acc1 = WMMA_F16(a, b1, acc1);
  }
  float ssum = 0.f;
#pragma unroll
  for (int c = 0; c < 2; ++c) {
    v8f acc = c ? acc1 : acc0;
    int n = n0 + (c << 4) + (lane & 15);
#pragma unroll
    for (int j = 0; j < 8; ++j) {
      int m = m0 + j + ((lane >> 4) << 3);
      int idx = m * D_IN + n;
      float ih = 1.f / (1.f + __expf(-acc[j]));
      IhatOut[idx] = ih;
      float d = It[idx] - ih;
      ssum += d * d;
    }
  }
  ssum = wave_sum(ssum);
  if (lane == 0) atomicAdd(&loss[0], ssum);
}

// ---------------------------------------------------------------------------
// K7: finalize means
// ---------------------------------------------------------------------------
__global__ void finalize_kernel(float* __restrict__ loss) {
  if (threadIdx.x == 0 && blockIdx.x == 0) {
    loss[0] *= (1.f / ((float)BATCH * (float)D_IN));
    loss[1] *= (1.f / ((float)BATCH * (float)D_Z));
    loss[2] *= (1.f / ((float)BATCH * (float)D_Z));
  }
}

// ---------------------------------------------------------------------------
// Host launcher
// ---------------------------------------------------------------------------
extern "C" void kernel_launch(void* const* d_in, const int* in_sizes, int n_in,
                              void* d_out, int out_size, void* d_ws, size_t ws_size,
                              hipStream_t stream) {
  (void)in_sizes; (void)n_in; (void)out_size; (void)ws_size;

  const float* I_t      = (const float*)d_in[0];
  const float* h_m_1    = (const float*)d_in[1];
  const float* z_m_1    = (const float*)d_in[2];
  const float* theta_m1 = (const float*)d_in[3];
  const float* eps_z    = (const float*)d_in[4];
  const float* eps_zhat = (const float*)d_in[5];
  const float* W_post_mu = (const float*)d_in[6];
  const float* W_post_lv = (const float*)d_in[7];
  const float* W_z2h     = (const float*)d_in[8];
  const float* W_h2h     = (const float*)d_in[9];
  const float* W_prior_mu = (const float*)d_in[10];
  const float* W_prior_lv = (const float*)d_in[11];
  const float* W_i2t      = (const float*)d_in[12];
  const float* W_vip2t    = (const float*)d_in[13];
  const float* W_t2z      = (const float*)d_in[14];
  const float* W_rec1     = (const float*)d_in[15];
  const float* W_rec2     = (const float*)d_in[16];

  // d_out layout: I_hat, z, h, z_hat, [spatial, temporal, energy]
  float* out      = (float*)d_out;
  float* IhatOut  = out;
  float* zOut     = out + (size_t)BATCH * D_IN;
  float* hOut     = zOut + (size_t)BATCH * D_Z;
  float* zhatOut  = hOut + (size_t)BATCH * D_H;
  float* lossOut  = zhatOut + (size_t)BATCH * D_Z;

  // Workspace (f16) layout
  _Float16* w = (_Float16*)d_ws;
  size_t off = 0;
  _Float16* It_h   = w + off; off += (size_t)BATCH * D_IN;
  _Float16* Hm1_h  = w + off; off += (size_t)BATCH * D_H;
  _Float16* Zm1_h  = w + off; off += (size_t)BATCH * D_Z;
  _Float16* Sigp_h = w + off; off += (size_t)BATCH * D_Z;
  _Float16* Th16   = w + off; off += (size_t)BATCH * 32;     // padded theta
  _Float16* Z_h    = w + off; off += (size_t)BATCH * D_Z;
  _Float16* Zr_h   = w + off; off += (size_t)BATCH * D_REC;
  _Float16* Wpm_h  = w + off; off += (size_t)D_Z * D_IN;
  _Float16* Wpl_h  = w + off; off += (size_t)D_Z * D_IN;
  _Float16* Wz2h_h = w + off; off += (size_t)D_H * D_Z;
  _Float16* Wh2h_h = w + off; off += (size_t)D_H * D_H;
  _Float16* Wprm_h = w + off; off += (size_t)D_Z * D_H;
  _Float16* Wprl_h = w + off; off += (size_t)D_Z * D_H;
  _Float16* Wi2t_h = w + off; off += (size_t)D_T * D_IN;
  _Float16* Wvip_h = w + off; off += (size_t)D_T * D_Z;
  _Float16* Wt2z_h = w + off; off += (size_t)D_Z * 32;       // padded
  _Float16* Wr1_h  = w + off; off += (size_t)D_REC * D_Z;
  _Float16* Wr2_h  = w + off; off += (size_t)D_IN * D_REC;

  // Zero the three loss accumulators (graph-capture-legal)
  hipMemsetAsync((void*)lossOut, 0, 3 * sizeof(float), stream);

  const int T = 256;
  auto cvt = [&](const float* s, _Float16* d, int n, int relu) {
    cvt_f16_kernel<<<(n + T - 1) / T, T, 0, stream>>>(s, d, n, relu);
  };
  cvt(I_t,        It_h,   BATCH * D_IN, 0);
  cvt(h_m_1,      Hm1_h,  BATCH * D_H,  0);
  cvt(z_m_1,      Zm1_h,  BATCH * D_Z,  0);
  cvt(W_post_mu,  Wpm_h,  D_Z * D_IN,   0);
  cvt(W_post_lv,  Wpl_h,  D_Z * D_IN,   0);
  cvt(W_z2h,      Wz2h_h, D_H * D_Z,    0);
  cvt(W_h2h,      Wh2h_h, D_H * D_H,    0);
  cvt(W_prior_mu, Wprm_h, D_Z * D_H,    0);
  cvt(W_prior_lv, Wprl_h, D_Z * D_H,    0);
  cvt(W_i2t,      Wi2t_h, D_T * D_IN,   0);
  cvt(W_vip2t,    Wvip_h, D_T * D_Z,    1);   // Positive parametrization
  cvt(W_rec1,     Wr1_h,  D_REC * D_Z,  0);
  cvt(W_rec2,     Wr2_h,  D_IN * D_REC, 0);
  t2z_pad_kernel<<<(D_Z * 32 + T - 1) / T, T, 0, stream>>>(W_t2z, Wt2z_h);

  // 8 waves per 256-thread block; each wave owns a 16x32 output tile.
  const int tilesZ2  = (BATCH / 16) * (D_Z / 32);    // 4096
  const int tilesT   = (BATCH / 16);                 // 512
  const int tilesIn2 = (BATCH / 16) * (D_IN / 32);   // 16384

  prior_kernel<<<tilesZ2 / 8, 256, 0, stream>>>(Hm1_h, Wprm_h, Wprl_h,
                                                eps_zhat, zhatOut, Sigp_h);
  hstate_kernel<<<tilesZ2 / 8, 256, 0, stream>>>(Zm1_h, Hm1_h, Wz2h_h, Wh2h_h, hOut);
  theta_kernel<<<tilesT / 8, 256, 0, stream>>>(It_h, Wi2t_h, Sigp_h, Wvip_h,
                                               theta_m1, Th16);
  // 128 KB dynamic LDS for the TDM-staged weight panels (harmless if the
  // device pass fell back to global loads).
  posterior_kernel<<<tilesZ2 / 8, 256, 131072, stream>>>(It_h, Wpm_h, Wpl_h, Th16,
                                                         Wt2z_h, eps_z, zhatOut,
                                                         zOut, Z_h, lossOut);
  rec1_kernel<<<tilesZ2 / 8, 256, 0, stream>>>(Z_h, Wr1_h, Zr_h);
  rec2_kernel<<<tilesIn2 / 8, 256, 0, stream>>>(Zr_h, Wr2_h, I_t, IhatOut, lossOut);
  finalize_kernel<<<1, 1, 0, stream>>>(lossOut);
}
